// GraphNet_19138374271129
// MI455X (gfx1250) — compile-verified, matrix operand order checked by reading, not run on hardware
//
#include <hip/hip_runtime.h>

// ---------------------------------------------------------------------------
// Problem constants (match reference)
// ---------------------------------------------------------------------------
#define BB    32          // graphs
#define NG0   1024        // nodes/graph at layer 0
#define FEAT_ 1036
#define KPAD0 1056        // FEAT_ rounded up to multiple of 32
#define NH    128
#define GD    32
#define P_    8
#define N0    (BB * NG0)  // 32768
#define E_    (N0 * 8)    // 262144

typedef __bf16 bf16x16 __attribute__((ext_vector_type(16)));
typedef float  f32x8   __attribute__((ext_vector_type(8)));
typedef unsigned int u32x4 __attribute__((ext_vector_type(4)));
typedef unsigned int u32x8 __attribute__((ext_vector_type(8)));

union FragU { unsigned int u[8]; bf16x16 v; };

__device__ __forceinline__ unsigned short f32_to_bf16(float f) {
    unsigned int u = __float_as_uint(f);
    u += 0x7FFFu + ((u >> 16) & 1u);       // round-to-nearest-even
    return (unsigned short)(u >> 16);
}

__device__ __forceinline__ unsigned int f32_ord(float f) {
    unsigned int u = __float_as_uint(f);
    return (u & 0x80000000u) ? ~u : (u | 0x80000000u);
}
__device__ __forceinline__ float f32_unord(unsigned int u) {
    return __uint_as_float((u & 0x80000000u) ? (u & 0x7FFFFFFFu) : ~u);
}

// ---------------------------------------------------------------------------
// Fills
// ---------------------------------------------------------------------------
__global__ void fill_f32_kernel(float* p, float v, int n) {
    for (int i = blockIdx.x * blockDim.x + threadIdx.x; i < n;
         i += gridDim.x * blockDim.x) p[i] = v;
}
__global__ void fill_i32_kernel(int* p, int v, int n) {
    for (int i = blockIdx.x * blockDim.x + threadIdx.x; i < n;
         i += gridDim.x * blockDim.x) p[i] = v;
}

// ---------------------------------------------------------------------------
// Column max of x[:, :12] (order-preserving uint encoding + atomicMax)
// ---------------------------------------------------------------------------
__global__ void colmax_kernel(const float* __restrict__ x, unsigned int* bits, int n) {
    unsigned int loc[12];
#pragma unroll
    for (int j = 0; j < 12; ++j) loc[j] = 0u;
    for (int i = blockIdx.x * blockDim.x + threadIdx.x; i < n;
         i += gridDim.x * blockDim.x) {
#pragma unroll
        for (int j = 0; j < 12; ++j) {
            unsigned int o = f32_ord(x[(long)i * FEAT_ + j]);
            if (o > loc[j]) loc[j] = o;
        }
    }
#pragma unroll
    for (int j = 0; j < 12; ++j) atomicMax(&bits[j], loc[j]);
}
__global__ void inv_scale_kernel(const unsigned int* bits, float* inv) {
    int j = threadIdx.x;
    if (j < 12) inv[j] = 1.0f / f32_unord(bits[j]);
}

// ---------------------------------------------------------------------------
// Weight prep: W[K][128] fp32 (row-major) -> Wt[128][Kpad] bf16 (transposed,
// zero-padded along K). One-shot per call; consumed by the TDM B-tile loads.
// ---------------------------------------------------------------------------
__global__ void prep_weight_kernel(const float* __restrict__ W,
                                   unsigned short* __restrict__ Wt,
                                   int K, int Kpad) {
    int tot = NH * Kpad;
    for (int i = blockIdx.x * blockDim.x + threadIdx.x; i < tot;
         i += gridDim.x * blockDim.x) {
        int c = i / Kpad, k = i % Kpad;
        float v = (k < K) ? W[(long)k * NH + c] : 0.0f;
        Wt[i] = f32_to_bf16(v);
    }
}

// ---------------------------------------------------------------------------
// bf16 WMMA GEMM:  C[M x 128] = A[M x K] * Bt^T, fp32 A, bf16 MACs.
//  - A tile (64x32) staged by shader with fp32->bf16 convert (+ column scale).
//  - B tile (32x128) staged by the Tensor Data Mover: one tensor_load_to_lds
//    per K-step, TDM pad feature creates the 80B row stride in LDS.
// Block tile 64x128, 4 waves, each wave -> one 16-row strip, 8 accum tiles.
// ---------------------------------------------------------------------------
__global__ __launch_bounds__(128) void gemm_bf16_wmma_kernel(
    const float* __restrict__ A, int lda,
    const unsigned short* __restrict__ Bt,   // [128][Kpad] bf16, transposed
    float* __restrict__ C, int ldc,
    int M, int K, int Kpad,
    const float* __restrict__ colscale, int nscale)
{
    __shared__ unsigned short ldsA[64][34];   // [row][k], +2 pad
    __shared__ unsigned short ldsB[128][40];  // [col][k], 80B stride (TDM pad)

    const int tid  = threadIdx.x;
    const int lane = tid & 31;
    const int wave = tid >> 5;     // 0..3
    const int half = lane >> 4;    // 0/1
    const int l16  = lane & 15;

    const int blockM = blockIdx.x * 64;

    // Block-uniform TDM descriptor pieces (group1 constant across K-steps)
    const unsigned ldsb_off = (unsigned)(unsigned long long)(void*)&ldsB[0][0];
    u32x8 g1;
    g1[0] = (1u << 16)        // data_size = 2 bytes
          | (1u << 20)        // pad_enable
          | (3u << 22)        // pad_interval: 16 DWORDs (= one 64B tile row)
          | (3u << 25);       // pad_amount:   4 DWORDs (= 16B)
    g1[1] = ((unsigned)Kpad & 0xFFFFu) << 16;       // tensor_dim0[15:0]
    g1[2] = (((unsigned)Kpad >> 16) & 0xFFFFu)      // tensor_dim0[31:16]
          | (128u << 16);                           // tensor_dim1[15:0] = 128
    g1[3] = (32u << 16);                            // tile_dim0 = 32
    g1[4] = 128u;                                   // tile_dim1 = 128
    g1[5] = (unsigned)Kpad;                         // tensor_dim0_stride
    g1[6] = 0u;
    g1[7] = 0u;
    u32x4 gz = {0u, 0u, 0u, 0u};                    // groups 2/3 (2-D tile)

    f32x8 acc[8] = {};

    for (int k0 = 0; k0 < Kpad; k0 += 32) {
        // ---- stage A tile 64x32 (fp32 -> bf16, optional column scaling)
#pragma unroll
        for (int i = 0; i < 16; ++i) {
            int idx = i * 128 + tid;
            int r = idx >> 5, k = idx & 31;
            int gr = blockM + r, gk = k0 + k;
            float v = 0.0f;
            if (gr < M && gk < K) {
                v = A[(long)gr * lda + gk];
                if (gk < nscale) v *= colscale[gk];
            }
            ldsA[r][k] = f32_to_bf16(v);
        }

        // ---- stage B tile 32x128 via Tensor Data Mover (wave 0 issues)
        if (wave == 0) {
            unsigned long long gaddr =
                (unsigned long long)(const void*)Bt + (unsigned long long)k0 * 2ull;
            u32x4 g0;
            g0[0] = 1u;                                   // count = 1 descriptor
            g0[1] = ldsb_off;                             // lds_addr
            g0[2] = (unsigned)(gaddr & 0xFFFFFFFFull);    // global_addr[31:0]
            g0[3] = (unsigned)((gaddr >> 32) & 0x01FFFFFFull)
                  | (2u << 30);                           // addr[56:32] | type=2
            asm volatile("tensor_load_to_lds %0, %1, %2, %3"
                         :
                         : "s"(g0), "s"(g1), "s"(gz), "s"(gz)
                         : "memory");
            __builtin_amdgcn_s_wait_tensorcnt(0);
        }
        __syncthreads();   // A ds_stores + TDM completion visible to all waves

        // ---- A fragment: 16x32 bf16 per the CDNA5 wave32 layout
        FragU fa;
        int arow = wave * 16 + l16;
#pragma unroll
        for (int w = 0; w < 8; ++w) {
            int kp = (w < 4 ? 2 * w : 2 * w + 8) + 8 * half; // even -> 4B aligned
            fa.u[w] = *(const unsigned int*)&ldsA[arow][kp];
        }
        // ---- preload all 8 B fragments, then chain WMMAs (deep DS/WMMA overlap)
        FragU fb[8];
#pragma unroll
        for (int s = 0; s < 8; ++s) {
            int bcol = s * 16 + l16;
#pragma unroll
            for (int w = 0; w < 8; ++w) {
                int kp = (w < 4 ? 2 * w : 2 * w + 8) + 8 * half;
                fb[s].u[w] = *(const unsigned int*)&ldsB[bcol][kp];
            }
        }
#pragma unroll
        for (int s = 0; s < 8; ++s) {
            acc[s] = __builtin_amdgcn_wmma_f32_16x16x32_bf16(
                false, fa.v, false, fb[s].v, (short)0, acc[s], false, false);
        }
        __syncthreads();
    }

    // ---- store C per the 16x16 f32 C/D layout: VGPR r -> row (r + 8*half)
    int rowBase = blockM + wave * 16 + 8 * half;
#pragma unroll
    for (int s = 0; s < 8; ++s) {
        int col = s * 16 + l16;
#pragma unroll
        for (int r = 0; r < 8; ++r) {
            int row = rowBase + r;
            if (row < M) C[(long)row * ldc + col] = acc[s][r];
        }
    }
}

// ---------------------------------------------------------------------------
// Edge init / feature aggregation / scalar aggregation / remap
// ---------------------------------------------------------------------------
__global__ void edge_init_kernel(const int* __restrict__ eidx, int* src, int* dst,
                                 float* em, int E) {
    for (int i = blockIdx.x * blockDim.x + threadIdx.x; i < E;
         i += gridDim.x * blockDim.x) {
        src[i] = eidx[i];
        dst[i] = eidx[E + i];
        em[i]  = 1.0f;
    }
}

// wave-per-edge scatter-add of 128-dim projected features
__global__ void edge_agg_kernel(const int* __restrict__ src, const int* __restrict__ dst,
                                const float* __restrict__ em,
                                const float* __restrict__ y,
                                float* agg, float* cnt, int E) {
    int gt = blockIdx.x * blockDim.x + threadIdx.x;
    int w = gt >> 5, lane = gt & 31;
    int nw = (gridDim.x * blockDim.x) >> 5;
    for (int e = w; e < E; e += nw) {
        if (em[e] > 0.0f) {
            int s = src[e], d = dst[e];
            __builtin_prefetch(&y[(long)s * NH + lane], 0, 0);
#pragma unroll
            for (int j = 0; j < 4; ++j) {
                int c = lane + 32 * j;
                atomicAdd(&agg[(long)d * NH + c], y[(long)s * NH + c]);
            }
            if (lane == 0) atomicAdd(&cnt[d], 1.0f);
        }
    }
}

__global__ void combine_kernel(const float* __restrict__ agg, const float* __restrict__ cnt,
                               const float* __restrict__ yr, const float* __restrict__ bl,
                               float* xout, int n) {
    long tot = (long)n * NH;
    for (long i = (long)blockIdx.x * blockDim.x + threadIdx.x; i < tot;
         i += (long)gridDim.x * blockDim.x) {
        int node = (int)(i >> 7), c = (int)(i & 127);
        float cn = cnt[node]; cn = cn < 1.0f ? 1.0f : cn;
        float v = agg[i] / cn + bl[c] + yr[i];
        xout[i] = fmaxf(v, 0.0f);
    }
}

// wave-per-node dual 128-dot: t = x.Ws_rel, r = x.Ws_root
__global__ void proj_score_kernel(const float* __restrict__ xin,
                                  const float* __restrict__ wrel,
                                  const float* __restrict__ wroot,
                                  float* t_out, float* r_out, int n) {
    int gt = blockIdx.x * blockDim.x + threadIdx.x;
    int w = gt >> 5, lane = gt & 31;
    int nw = (gridDim.x * blockDim.x) >> 5;
    for (int i = w; i < n; i += nw) {
        float s1 = 0.0f, s2 = 0.0f;
#pragma unroll
        for (int j = 0; j < 4; ++j) {
            int c = lane + 32 * j;
            float v = xin[(long)i * NH + c];
            s1 += v * wrel[c];
            s2 += v * wroot[c];
        }
#pragma unroll
        for (int o = 16; o > 0; o >>= 1) {
            s1 += __shfl_xor(s1, o, 32);
            s2 += __shfl_xor(s2, o, 32);
        }
        if (lane == 0) { t_out[i] = s1; r_out[i] = s2; }
    }
}

__global__ void sagg_kernel(const int* __restrict__ src, const int* __restrict__ dst,
                            const float* __restrict__ em, const float* __restrict__ t,
                            float* sagg, int E) {
    for (int i = blockIdx.x * blockDim.x + threadIdx.x; i < E;
         i += gridDim.x * blockDim.x)
        if (em[i] > 0.0f) atomicAdd(&sagg[dst[i]], t[src[i]]);
}

__global__ void score_kernel(const float* __restrict__ sagg, const float* __restrict__ r,
                             const float* __restrict__ bs, float* score, int n) {
    for (int i = blockIdx.x * blockDim.x + threadIdx.x; i < n;
         i += gridDim.x * blockDim.x)
        score[i] = sagg[i] + bs[0] + r[i];
}

// ---------------------------------------------------------------------------
// Per-graph top-k via in-LDS bitonic sort (descending). ngPad = pow2 >= ng.
// ---------------------------------------------------------------------------
__global__ void topk_kernel(const float* __restrict__ score, int* perm,
                            int ng, int ngPad, int k) {
    __shared__ float sc[1024];
    __shared__ int   ix[1024];
    int g = blockIdx.x;
    for (int j = threadIdx.x; j < ngPad; j += blockDim.x) {
        sc[j] = (j < ng) ? score[g * ng + j] : -3.4e38f;
        ix[j] = j;
    }
    __syncthreads();
    for (int size = 2; size <= ngPad; size <<= 1) {
        for (int stride = size >> 1; stride > 0; stride >>= 1) {
            for (int i = threadIdx.x; i < ngPad; i += blockDim.x) {
                int j = i ^ stride;
                if (j > i) {
                    bool descBlk = ((i & size) == 0);
                    bool sw = descBlk ? (sc[i] < sc[j]) : (sc[i] > sc[j]);
                    if (sw) {
                        float tf = sc[i]; sc[i] = sc[j]; sc[j] = tf;
                        int   ti = ix[i]; ix[i] = ix[j]; ix[j] = ti;
                    }
                }
            }
            __syncthreads();
        }
    }
    for (int j = threadIdx.x; j < k; j += blockDim.x)
        perm[g * k + j] = g * ng + ix[j];
}

__global__ void nodemap_set_kernel(int* nmap, const int* __restrict__ perm, int m) {
    for (int i = blockIdx.x * blockDim.x + threadIdx.x; i < m;
         i += gridDim.x * blockDim.x) nmap[perm[i]] = i;
}

__global__ void gather_kernel(const float* __restrict__ xin, const float* __restrict__ score,
                              const int* __restrict__ perm, float* xout, int nNew) {
    long tot = (long)nNew * NH;
    for (long i = (long)blockIdx.x * blockDim.x + threadIdx.x; i < tot;
         i += (long)gridDim.x * blockDim.x) {
        int j = (int)(i >> 7), c = (int)(i & 127);
        int p = perm[j];
        xout[i] = xin[(long)p * NH + c] * tanhf(score[p]);
    }
}

__global__ void edge_remap_kernel(int* src, int* dst, float* em,
                                  const int* __restrict__ nmap, int E) {
    for (int i = blockIdx.x * blockDim.x + threadIdx.x; i < E;
         i += gridDim.x * blockDim.x) {
        int ns = nmap[src[i]], nd = nmap[dst[i]];
        bool ok = (ns >= 0) && (nd >= 0);
        em[i]  = ok ? em[i] : 0.0f;
        src[i] = ok ? ns : 0;
        dst[i] = ok ? nd : 0;
    }
}

// per-graph [gmp | gap] readout accumulated across layers
__global__ void readout_kernel(const float* __restrict__ xp, int ng, float* xs_acc) {
    int g = blockIdx.x, c = threadIdx.x;   // 128 threads
    float mx = -3.4e38f, sm = 0.0f;
    for (int j = 0; j < ng; ++j) {
        float v = xp[((long)(g * ng + j)) * NH + c];
        mx = fmaxf(mx, v);
        sm += v;
    }
    xs_acc[g * 256 + c]       += mx;
    xs_acc[g * 256 + 128 + c] += sm / (float)ng;
}

// ---------------------------------------------------------------------------
// Head: encoder MLP, patient mean, grade log-softmax, hazard
// ---------------------------------------------------------------------------
__global__ void head1_kernel(const float* __restrict__ xs, const float* __restrict__ W1,
                             const float* __restrict__ b1, float* h1) {
    int g = blockIdx.x, c = threadIdx.x;   // 128
    float s = b1[c];
    for (int i = 0; i < 256; ++i) s += xs[g * 256 + i] * W1[i * NH + c];
    h1[g * NH + c] = fmaxf(s, 0.0f);
}

__global__ void head2_kernel(const float* __restrict__ h1, const float* __restrict__ W2,
                             const float* __restrict__ b2, const int* __restrict__ pat,
                             float* fsum, float* fcnt) {
    int g = blockIdx.x, c = threadIdx.x;   // 32
    float s = b2[c];
    for (int i = 0; i < NH; ++i) s += h1[g * NH + i] * W2[i * GD + c];
    s = fmaxf(s, 0.0f);
    int p = pat[g];
    atomicAdd(&fsum[p * GD + c], s);
    if (c == 0) atomicAdd(&fcnt[p], 1.0f);
}

__global__ void final_kernel(const float* __restrict__ fsum, const float* __restrict__ fcnt,
                             const float* __restrict__ gw, const float* __restrict__ gb,
                             const float* __restrict__ hw, const float* __restrict__ hb,
                             float* out) {
    __shared__ float feat[P_ * GD];
    int t = threadIdx.x;                   // 256 threads
    if (t < P_ * GD) {
        int p = t / GD;
        float c = fcnt[p]; c = c < 1.0f ? 1.0f : c;
        float v = fsum[t] / c;
        feat[t] = v;
        out[t] = v;                        // features [P,32]
    }
    __syncthreads();
    if (t < P_) {
        float l[3];
#pragma unroll
        for (int j = 0; j < 3; ++j) {
            float s = gb[j];
            for (int i = 0; i < GD; ++i) s += feat[t * GD + i] * gw[i * 3 + j];
            l[j] = s;
        }
        float m = fmaxf(l[0], fmaxf(l[1], l[2]));
        float lse = logf(expf(l[0] - m) + expf(l[1] - m) + expf(l[2] - m)) + m;
#pragma unroll
        for (int j = 0; j < 3; ++j) out[P_ * GD + t * 3 + j] = l[j] - lse;  // grade
        float z = hb[0];
        for (int i = 0; i < GD; ++i) z += feat[t * GD + i] * hw[i];
        out[P_ * GD + P_ * 3 + t] = 6.0f / (1.0f + expf(-z)) - 3.0f;        // hazard
    }
}

// ---------------------------------------------------------------------------
// Host orchestration (all launches on `stream`; graph-capture safe)
// ---------------------------------------------------------------------------
extern "C" void kernel_launch(void* const* d_in, const int* in_sizes, int n_in,
                              void* d_out, int out_size, void* d_ws, size_t ws_size,
                              hipStream_t stream) {
    (void)in_sizes; (void)n_in; (void)out_size; (void)ws_size;

    // JAX pytree flatten order (dict keys sorted; ASCII: 'W' < 'b')
    const float* x      = (const float*)d_in[0];
    const int*   eidx   = (const int*)d_in[1];
    const int*   pat    = (const int*)d_in[3];
    const float* enc_W1 = (const float*)d_in[4];
    const float* enc_W2 = (const float*)d_in[5];
    const float* enc_b1 = (const float*)d_in[6];
    const float* enc_b2 = (const float*)d_in[7];
    const float* gradeW = (const float*)d_in[8];
    const float* gradeB = (const float*)d_in[9];
    const float* hazW   = (const float*)d_in[10];
    const float* hazB   = (const float*)d_in[11];
    // per layer L (base 12 + 6L): Wl, Wr, Ws_rel, Ws_root, bl, bs_rel

    char* ws = (char*)d_ws;
    size_t off = 0;
    auto take = [&](size_t bytes) -> char* {
        char* p = ws + off;
        off += (bytes + 255) & ~(size_t)255;
        return p;
    };
    unsigned int* colmax = (unsigned int*)take(12 * 4);
    float* inv    = (float*)take(12 * 4);
    float* xConv  = (float*)take((size_t)N0 * NH * 4);
    float* xPool  = (float*)take((size_t)N0 * NH * 4);
    float* yl     = (float*)take((size_t)N0 * NH * 4);
    float* yr     = (float*)take((size_t)N0 * NH * 4);
    float* agg    = (float*)take((size_t)N0 * NH * 4);
    float* cnt    = (float*)take((size_t)N0 * 4);
    float* tbuf   = (float*)take((size_t)N0 * 4);
    float* rbuf   = (float*)take((size_t)N0 * 4);
    float* saggb  = (float*)take((size_t)N0 * 4);
    float* scoreb = (float*)take((size_t)N0 * 4);
    int*   srcb   = (int*)take((size_t)E_ * 4);
    int*   dstb   = (int*)take((size_t)E_ * 4);
    float* emaskb = (float*)take((size_t)E_ * 4);
    int*   permb  = (int*)take((size_t)BB * 205 * 4);
    int*   nmap   = (int*)take((size_t)N0 * 4);
    float* xsacc  = (float*)take((size_t)BB * 256 * 4);
    float* h1     = (float*)take((size_t)BB * NH * 4);
    float* fsum   = (float*)take((size_t)P_ * GD * 4);
    float* fcnt   = (float*)take((size_t)P_ * 4);
    // transposed bf16 weights for the TDM-fed GEMM B operand
    unsigned short* Wt[3][2];
    const int kpadL[3] = {KPAD0, NH, NH};
    for (int L = 0; L < 3; ++L)
        for (int s = 0; s < 2; ++s)
            Wt[L][s] = (unsigned short*)take((size_t)NH * kpadL[L] * 2);

    const int ngs[3]   = {1024, 205, 41};
    const int ks[3]    = {205, 41, 9};
    const int ngPad[3] = {1024, 256, 64};
    const int kL[3]    = {FEAT_, NH, NH};

    // --- normalization scales for x[:, :12]
    fill_i32_kernel<<<1, 32, 0, stream>>>((int*)colmax, 0, 12);
    colmax_kernel<<<256, 256, 0, stream>>>(x, colmax, N0);
    inv_scale_kernel<<<1, 32, 0, stream>>>(colmax, inv);

    // --- weight prep (transpose + bf16 + K padding)
    for (int L = 0; L < 3; ++L) {
        prep_weight_kernel<<<256, 256, 0, stream>>>(
            (const float*)d_in[12 + 6 * L + 0], Wt[L][0], kL[L], kpadL[L]);
        prep_weight_kernel<<<256, 256, 0, stream>>>(
            (const float*)d_in[12 + 6 * L + 1], Wt[L][1], kL[L], kpadL[L]);
    }

    // --- edges + accumulators
    edge_init_kernel<<<512, 256, 0, stream>>>(eidx, srcb, dstb, emaskb, E_);
    fill_f32_kernel<<<64, 256, 0, stream>>>(xsacc, 0.0f, BB * 256);

    for (int L = 0; L < 3; ++L) {
        const float* Wsr  = (const float*)d_in[12 + 6 * L + 2];
        const float* Wsro = (const float*)d_in[12 + 6 * L + 3];
        const float* bl   = (const float*)d_in[12 + 6 * L + 4];
        const float* bsr  = (const float*)d_in[12 + 6 * L + 5];

        const int n    = BB * ngs[L];
        const int nNew = BB * ks[L];
        dim3 gg((n + 63) / 64, 1);

        // 1) project: yl = x @ Wl, yr = x @ Wr  (bf16 WMMA + TDM B staging)
        const float* Ain = (L == 0) ? x : xPool;
        const int lda    = (L == 0) ? FEAT_ : NH;
        const int nsc    = (L == 0) ? 12 : 0;
        gemm_bf16_wmma_kernel<<<gg, 128, 0, stream>>>(Ain, lda, Wt[L][0], yl, NH,
                                                      n, kL[L], kpadL[L], inv, nsc);
        gemm_bf16_wmma_kernel<<<gg, 128, 0, stream>>>(Ain, lda, Wt[L][1], yr, NH,
                                                      n, kL[L], kpadL[L], inv, nsc);

        // 2) mean-aggregate projected neighbors
        fill_f32_kernel<<<1024, 256, 0, stream>>>(agg, 0.0f, n * NH);
        fill_f32_kernel<<<128, 256, 0, stream>>>(cnt, 0.0f, n);
        edge_agg_kernel<<<1024, 256, 0, stream>>>(srcb, dstb, emaskb, yl, agg, cnt, E_);

        // 3) x_new = relu(agg/max(cnt,1) + bl + yr)
        combine_kernel<<<1024, 256, 0, stream>>>(agg, cnt, yr, bl, xConv, n);

        // 4) scores: project first, then sum-aggregate scalars
        proj_score_kernel<<<512, 256, 0, stream>>>(xConv, Wsr, Wsro, tbuf, rbuf, n);
        fill_f32_kernel<<<128, 256, 0, stream>>>(saggb, 0.0f, n);
        sagg_kernel<<<512, 256, 0, stream>>>(srcb, dstb, emaskb, tbuf, saggb, E_);
        score_kernel<<<128, 256, 0, stream>>>(saggb, rbuf, bsr, scoreb, n);

        // 5) per-graph top-k, gather with tanh gating, remap edges
        topk_kernel<<<BB, 512, 0, stream>>>(scoreb, permb, ngs[L], ngPad[L], ks[L]);
        fill_i32_kernel<<<128, 256, 0, stream>>>(nmap, -1, n);
        nodemap_set_kernel<<<32, 256, 0, stream>>>(nmap, permb, nNew);
        gather_kernel<<<512, 256, 0, stream>>>(xConv, scoreb, permb, xPool, nNew);
        edge_remap_kernel<<<512, 256, 0, stream>>>(srcb, dstb, emaskb, nmap, E_);

        // 6) readout [gmp | gap], accumulated across layers
        readout_kernel<<<BB, 128, 0, stream>>>(xPool, ks[L], xsacc);
    }

    // --- head MLP + patient mean + outputs
    head1_kernel<<<BB, 128, 0, stream>>>(xsacc, enc_W1, enc_b1, h1);
    fill_f32_kernel<<<1, 256, 0, stream>>>(fsum, 0.0f, P_ * GD);
    fill_f32_kernel<<<1, 32, 0, stream>>>(fcnt, 0.0f, P_);
    head2_kernel<<<BB, 32, 0, stream>>>(h1, enc_W2, enc_b2, pat, fsum, fcnt);
    final_kernel<<<1, 256, 0, stream>>>(fsum, fcnt, gradeW, gradeB, hazW, hazB,
                                        (float*)d_out);
}
